// GCNBlock_46462956208753
// MI455X (gfx1250) — matrix operand order repaired and checked by reading, not compiled, not run on hardware
//
#include <hip/hip_runtime.h>

// ---------------------------------------------------------------------------
// GCN block: 3 x (GEMM -> gather/scatter-add -> finalize+residual)
// N=100000 nodes, E=1600000 edges, D=128 channels.
// GEMM uses CDNA5 WMMA bf16 (v_wmma_f32_16x16x32_bf16), edge phase uses
// float4 gathers + global_atomic_add_f32 scatters.
// ---------------------------------------------------------------------------

#define DCH 128

typedef __attribute__((ext_vector_type(16))) __bf16 v16bf;
typedef __attribute__((ext_vector_type(8)))  __bf16 v8bf;
typedef __attribute__((ext_vector_type(8)))  float  v8f;

// ---------------- normalization prep ----------------

__global__ void deg_init_kernel(float* __restrict__ deg, int n) {
    int i = blockIdx.x * blockDim.x + threadIdx.x;
    if (i < n) deg[i] = 2.0f;   // self-loop weight (improved=True)
}

__global__ void deg_acc_kernel(const int* __restrict__ dst, float* __restrict__ deg, int ne) {
    int e = blockIdx.x * blockDim.x + threadIdx.x;
    if (e < ne) atomicAdd(&deg[dst[e]], 1.0f);
}

__global__ void dinv_kernel(float* __restrict__ deg, int n) {
    int i = blockIdx.x * blockDim.x + threadIdx.x;
    if (i < n) {
        float v = deg[i];
        deg[i] = (v > 0.0f) ? rsqrtf(v) : 0.0f;   // in-place deg -> dinv
    }
}

__global__ void norm_kernel(const int* __restrict__ src, const int* __restrict__ dst,
                            const float* __restrict__ dinv, float* __restrict__ norm, int ne) {
    int e = blockIdx.x * blockDim.x + threadIdx.x;
    if (e < ne) norm[e] = dinv[src[e]] * dinv[dst[e]];   // edge weight 1.0
}

__global__ void zero_kernel(float4* __restrict__ p, int n4) {
    int i = blockIdx.x * blockDim.x + threadIdx.x;
    if (i < n4) p[i] = make_float4(0.f, 0.f, 0.f, 0.f);
}

// ---------------- dense transform: H = X * W  (WMMA bf16) ----------------
// Block: 256 threads = 8 waves. Output tile: 128 rows x 128 cols.
// LDS: X tile (bf16, row-major) + W (bf16, pre-swizzled into B-fragment order).

__global__ __launch_bounds__(256) void gemm_wmma_kernel(const float* __restrict__ X,
                                                        const float* __restrict__ W,
                                                        float* __restrict__ H, int nrows) {
    __shared__ __bf16 sX[DCH * DCH];   // 32 KB
    __shared__ __bf16 sW[DCH * DCH];   // 32 KB, swizzled

    const int tid = threadIdx.x;
    const int rowBase = blockIdx.x * DCH;

    // Stage X tile: fp32 -> bf16, coalesced float4 loads (rows past N -> 0).
    for (int i = tid; i < DCH * DCH / 4; i += 256) {
        const int r  = i >> 5;        // row within tile (32 float4 per row)
        const int c4 = i & 31;
        const int row = rowBase + r;
        float4 v = make_float4(0.f, 0.f, 0.f, 0.f);
        if (row < nrows) v = ((const float4*)(X + (size_t)row * DCH))[c4];
        __bf16* p = &sX[r * DCH + c4 * 4];
        p[0] = (__bf16)v.x; p[1] = (__bf16)v.y; p[2] = (__bf16)v.z; p[3] = (__bf16)v.w;
    }

    // Stage W swizzled into B-fragment lane order:
    //   fragment f = nt*4 + kt holds the 32x16 tile (k0=32*kt, n0=16*nt);
    //   lane L (L<16: K 0..15, L>=16: K 16..31 of the tile), element q = kk&15.
    for (int i = tid; i < DCH * DCH / 4; i += 256) {
        const int k  = i >> 5;        // W row (input channel)
        const int c4 = i & 31;
        const float4 v = ((const float4*)(W + (size_t)k * DCH))[c4];
        const float vv[4] = { v.x, v.y, v.z, v.w };
        const int kt = k >> 5, kk = k & 31;
        const int Lhi = (kk & 16);            // 0 or 16
        const int q   = kk & 15;
#pragma unroll
        for (int u = 0; u < 4; ++u) {
            const int n  = c4 * 4 + u;        // W col (output channel)
            const int nt = n >> 4;
            const int L  = Lhi | (n & 15);
            const int lidx = ((nt * 4 + kt) << 9) + (L << 4) + q;
            sW[lidx] = (__bf16)vv[u];
        }
    }
    __syncthreads();

    const int wave = tid >> 5;
    const int lane = tid & 31;
    const int m0   = wave << 4;                 // 16-row strip per wave
    const int arow = m0 + (lane & 15);
    const int ahalf = (lane >> 4) << 3;         // 0 or 8 (A-layout half)

    v8f acc[8] = {};                            // 8 col-tiles of 16x16 f32

#pragma unroll
    for (int kt = 0; kt < 4; ++kt) {
        const int k0 = kt << 5;
        // A fragment: contiguous k order (lane half h holds K 16h..16h+15), matching sW.
        const v8bf a0 = *(const v8bf*)&sX[arow * DCH + k0 + 2 * ahalf];
        const v8bf a1 = *(const v8bf*)&sX[arow * DCH + k0 + 2 * ahalf + 8];
        v16bf a;
#pragma unroll
        for (int t = 0; t < 8; ++t) { a[t] = a0[t]; a[t + 8] = a1[t]; }

#pragma unroll
        for (int nt = 0; nt < 8; ++nt) {
            const __bf16* bp = &sW[((nt * 4 + kt) << 9) + (lane << 4)];
            const v8bf b0 = *(const v8bf*)bp;
            const v8bf b1 = *(const v8bf*)(bp + 8);
            v16bf b;
#pragma unroll
            for (int t = 0; t < 8; ++t) { b[t] = b0[t]; b[t + 8] = b1[t]; }
            acc[nt] = __builtin_amdgcn_wmma_f32_16x16x32_bf16(
                false, a, false, b, (short)0, acc[nt], false, false);
        }
    }

    // Store: C/D layout — VGPR j: lanes 0-15 -> M=j, lanes 16-31 -> M=8+j.
    const int srow = rowBase + m0 + ((lane >> 4) << 3);
    const int scol = lane & 15;
#pragma unroll
    for (int nt = 0; nt < 8; ++nt) {
#pragma unroll
        for (int j = 0; j < 8; ++j) {
            const int row = srow + j;
            if (row < nrows) H[(size_t)row * DCH + nt * 16 + scol] = acc[nt][j];
        }
    }
}

// ---------------- edge gather + scatter-add ----------------
// One wave per edge: 32 lanes x float4 = 128 channels.

__global__ __launch_bounds__(256) void scatter_kernel(const float* __restrict__ h,
                                                      const int* __restrict__ src,
                                                      const int* __restrict__ dst,
                                                      const float* __restrict__ norm,
                                                      float* __restrict__ agg, int ne) {
    const int e = blockIdx.x * 8 + (threadIdx.x >> 5);
    if (e >= ne) return;
    const int lane = threadIdx.x & 31;
    const int s = __builtin_amdgcn_readfirstlane(src[e]);
    const int d = __builtin_amdgcn_readfirstlane(dst[e]);
    const float w = norm[e];
    const float4 v = ((const float4*)(h + (size_t)s * DCH))[lane];
    float* a = agg + (size_t)d * DCH + lane * 4;
    atomicAdd(a + 0, v.x * w);
    atomicAdd(a + 1, v.y * w);
    atomicAdd(a + 2, v.z * w);
    atomicAdd(a + 3, v.w * w);
}

// ---------------- finalize: out = agg + 2*dinv^2*h + b + resid (+ x) ----------------

__global__ void finalize_kernel(const float* __restrict__ agg, const float* __restrict__ h,
                                const float* __restrict__ dinv, const float* __restrict__ bias,
                                const float* __restrict__ resid, const float* __restrict__ gres,
                                float* __restrict__ out, int n_nodes) {
    const int i = blockIdx.x * blockDim.x + threadIdx.x;   // one float4 per thread
    const int total = n_nodes * (DCH / 4);
    if (i >= total) return;
    const int n  = i >> 5;
    const int c4 = i & 31;
    float s = dinv[n];
    s = 2.0f * s * s;                                      // self-loop: dinv * 2.0 * dinv
    const float4 a  = ((const float4*)agg)[i];
    const float4 hh = ((const float4*)h)[i];
    const float4 r  = ((const float4*)resid)[i];
    const float4 bb = ((const float4*)bias)[c4];
    float4 o;
    o.x = a.x + s * hh.x + bb.x + r.x;
    o.y = a.y + s * hh.y + bb.y + r.y;
    o.z = a.z + s * hh.z + bb.z + r.z;
    o.w = a.w + s * hh.w + bb.w + r.w;
    if (gres) {
        const float4 g = ((const float4*)gres)[i];
        o.x += g.x; o.y += g.y; o.z += g.z; o.w += g.w;
    }
    ((float4*)out)[i] = o;
}

// ---------------- driver ----------------

extern "C" void kernel_launch(void* const* d_in, const int* in_sizes, int n_in,
                              void* d_out, int out_size, void* d_ws, size_t ws_size,
                              hipStream_t stream) {
    const float* x  = (const float*)d_in[0];
    const int*   ei = (const int*)d_in[1];
    const float* Wm[3] = { (const float*)d_in[2], (const float*)d_in[4], (const float*)d_in[6] };
    const float* bm[3] = { (const float*)d_in[3], (const float*)d_in[5], (const float*)d_in[7] };
    float* out = (float*)d_out;

    const int Nn = in_sizes[0] / DCH;    // 100000
    const int Ee = in_sizes[1] / 2;      // 1600000
    const int* src = ei;
    const int* dst = ei + Ee;

    // Workspace carve-up (256B aligned slices).
    char* ws = (char*)d_ws;
    size_t off = 0;
    auto carve = [&](size_t bytes) -> char* {
        char* p = ws + off;
        off += (bytes + 255) & ~(size_t)255;
        return p;
    };
    float* dinv = (float*)carve((size_t)Nn * 4);
    float* norm = (float*)carve((size_t)Ee * 4);
    float* h    = (float*)carve((size_t)Nn * DCH * 4);
    float* agg  = (float*)carve((size_t)Nn * DCH * 4);
    float* cur  = (float*)carve((size_t)Nn * DCH * 4);
    (void)ws_size; (void)n_in; (void)out_size;

    const int B = 256;
    const int gN   = (Nn + B - 1) / B;
    const int gE   = (Ee + B - 1) / B;
    const int gGemm = (Nn + DCH - 1) / DCH;          // 782 blocks
    const int gScat = (Ee + 7) / 8;                  // 8 edges per block
    const int n4    = Nn * (DCH / 4);
    const int gElt  = (n4 + B - 1) / B;

    // Normalization: deg -> dinv -> per-edge norm.
    deg_init_kernel<<<gN, B, 0, stream>>>(dinv, Nn);
    deg_acc_kernel<<<gE, B, 0, stream>>>(dst, dinv, Ee);
    dinv_kernel<<<gN, B, 0, stream>>>(dinv, Nn);
    norm_kernel<<<gE, B, 0, stream>>>(src, dst, dinv, norm, Ee);

    const float* layer_in = x;
    for (int l = 0; l < 3; ++l) {
        zero_kernel<<<(n4 + B - 1) / B, B, 0, stream>>>((float4*)agg, n4);
        gemm_wmma_kernel<<<gGemm, B, 0, stream>>>(layer_in, Wm[l], h, Nn);
        scatter_kernel<<<gScat, B, 0, stream>>>(h, src, dst, norm, agg, Ee);
        const bool last = (l == 2);
        finalize_kernel<<<gElt, B, 0, stream>>>(agg, h, dinv, bm[l], layer_in,
                                                last ? x : nullptr,
                                                last ? out : cur, Nn);
        layer_in = cur;
    }
}
